// GraphAttention_36790689857540
// MI455X (gfx1250) — matrix operand order repaired and checked by reading, not compiled, not run on hardware
//
#include <hip/hip_runtime.h>
#include <hip/hip_bf16.h>

// ---------------------------------------------------------------------------
// Multi-head attention forward for MI455X (gfx1250, wave32).
//  - all GEMMs on V_WMMA_F32_16X16X32_BF16 (f32 accumulate)
//  - weight panels staged to LDS with the Tensor Data Mover
//    (tensor_load_to_lds + s_wait_tensorcnt), B-fragments from ds_load_b128
//  - flash-attention with online softmax, P relaid C->A through LDS
// ---------------------------------------------------------------------------

typedef __attribute__((ext_vector_type(16))) __bf16 v16bf;
typedef __attribute__((ext_vector_type(8)))  __bf16 v8bf;
typedef __attribute__((ext_vector_type(8)))  float  v8f;
typedef __attribute__((ext_vector_type(4)))  float  v4f;
typedef __attribute__((ext_vector_type(4)))  unsigned int v4u;
typedef __attribute__((ext_vector_type(8)))  int v8i;
typedef __attribute__((ext_vector_type(4)))  int v4i;

#define WMMA_BF16(a, b, c) \
  __builtin_amdgcn_wmma_f32_16x16x32_bf16(false, (a), false, (b), (short)0, (c), false, false)

#if __has_builtin(__builtin_amdgcn_tensor_load_to_lds) && \
    __has_builtin(__builtin_amdgcn_s_wait_tensorcnt)
#define USE_TDM 1
#else
#define USE_TDM 0
#endif

static constexpr int Bn = 2;          // batch
static constexpr int Tn = 2048;       // sequence
static constexpr int Dn = 768;        // model dim
static constexpr int Nh = 12;         // heads
static constexpr int Hd = 64;         // head dim
static constexpr int Mn = Bn * Tn;    // 4096 rows
static constexpr int NK = Nh * Hd;    // 768
static constexpr int KB = 256;        // K-chunk staged in LDS per panel

// ---------------------------------------------------------------------------
// Fragment loaders matching CDNA5 WMMA VGPR layouts (CDNA5 ISA §7.12.2):
//  A (16x32 bf16): lane<16 -> row=lane,    K = {k0..k0+7,  k0+16..k0+23}
//                  lane>=16 -> row=lane-16, K = {k0+8..k0+15, k0+24..k0+31}
//  B (32x16 bf16): lane<16 -> col=lane,    K = {k0..k0+7,  k0+16..k0+23}
//                  lane>=16 -> col=lane-16, K = {k0+8..k0+15, k0+24..k0+31}
// ---------------------------------------------------------------------------

__device__ inline v16bf a_frag_f32(const float* __restrict__ base, int ld,
                                   int row, int k0, int half) {
  const float* p = base + (size_t)row * ld + k0 + half * 8;
  v4f c0 = *(const v4f*)(p);
  v4f c1 = *(const v4f*)(p + 4);
  v4f c2 = *(const v4f*)(p + 16);
  v4f c3 = *(const v4f*)(p + 20);
  v16bf a;
#pragma unroll
  for (int i = 0; i < 4; ++i) {
    a[i]      = (__bf16)c0[i];
    a[4 + i]  = (__bf16)c1[i];
    a[8 + i]  = (__bf16)c2[i];
    a[12 + i] = (__bf16)c3[i];
  }
  return a;
}

__device__ inline v16bf a_frag_bf16(const __bf16* __restrict__ base, int ld,
                                    int row, int k0, int half) {
  const __bf16* p = base + (size_t)row * ld + k0 + half * 8;
  v8bf lo = *(const v8bf*)(p);
  v8bf hi = *(const v8bf*)(p + 16);
  return __builtin_shufflevector(lo, hi, 0, 1, 2, 3, 4, 5, 6, 7,
                                         8, 9, 10, 11, 12, 13, 14, 15);
}

// global, [col][K] layout with leading dim ld
__device__ inline v16bf b_frag_col(const __bf16* __restrict__ base, int ld,
                                   int col, int k0, int half) {
  const __bf16* p = base + (size_t)col * ld + k0 + half * 8;
  v8bf lo = *(const v8bf*)(p);
  v8bf hi = *(const v8bf*)(p + 16);
  return __builtin_shufflevector(lo, hi, 0, 1, 2, 3, 4, 5, 6, 7,
                                         8, 9, 10, 11, 12, 13, 14, 15);
}

// LDS panel, 16 cols x KB elements, row-contiguous
__device__ inline v16bf b_frag_lds(const __bf16* base, int col, int kk, int half) {
  const __bf16* p = base + col * KB + kk + half * 8;
  v8bf lo = *(const v8bf*)(p);
  v8bf hi = *(const v8bf*)(p + 16);
  return __builtin_shufflevector(lo, hi, 0, 1, 2, 3, 4, 5, 6, 7,
                                         8, 9, 10, 11, 12, 13, 14, 15);
}

// ---------------------------------------------------------------------------
// Tensor Data Mover: 2D tile (16 rows x KB elems, 2B/elem) global -> LDS.
// D# built per CDNA5 ISA.
// This toolchain's builtin takes 6 args: (g0, g1, g2, g3, g_extra8, cpol).
// ---------------------------------------------------------------------------
#if USE_TDM
__device__ inline void tdm_load_panel(const __bf16* gptr, unsigned lds_off) {
  unsigned long long ga = (unsigned long long)(uintptr_t)gptr;
  v4u g0;
  g0[0] = 1u;                                            // count=1, user desc
  g0[1] = lds_off;                                       // lds_addr
  g0[2] = (unsigned)ga;                                  // global_addr[31:0]
  g0[3] = ((unsigned)(ga >> 32) & 0x01FFFFFFu) | (2u << 30);  // addr[56:32]|type=2
  v8i g1;
  g1[0] = (int)(1u << 16);                               // data_size=1 (2B)
  g1[1] = (int)((KB & 0xFFFFu) << 16);                   // tensor_dim0[15:0]
  g1[2] = (int)((KB >> 16) | (16u << 16));               // dim0[31:16]|dim1[15:0]
  g1[3] = (int)((16u >> 16) | ((unsigned)KB << 16));     // dim1[31:16]|tile_dim0
  g1[4] = (int)16;                                       // tile_dim1=16, tile_dim2=0
  g1[5] = (int)Dn;                                       // tensor_dim0_stride[31:0]
  g1[6] = 0;
  g1[7] = 0;
  v4i z4 = {0, 0, 0, 0};
  v8i z8 = {0, 0, 0, 0, 0, 0, 0, 0};
  __builtin_amdgcn_tensor_load_to_lds(g0, g1, z4, z4, z8, 0);
}
#endif

// cooperative fallback: all 256 threads copy 16 x KB bf16 panel
__device__ inline void stage_panel_coop(const __bf16* __restrict__ g,
                                        __bf16* s, int ld) {
  for (int i = threadIdx.x; i < 16 * KB / 8; i += blockDim.x) {
    int col = i >> 5;            // KB/8 = 32 chunks per column
    int kk  = (i & 31) * 8;
    *(v8bf*)(s + col * KB + kk) = *(const v8bf*)(g + (size_t)col * ld + kk);
  }
}

// ---------------------------------------------------------------------------
// Kernel 1: f32 [rows][cols] -> bf16 transposed [cols][rows]
// ---------------------------------------------------------------------------
__global__ void transpose_bf16(const float* __restrict__ src,
                               __bf16* __restrict__ dst, int rows, int cols) {
  int i = blockIdx.x * blockDim.x + threadIdx.x;
  if (i < rows * cols) {
    int r = i / cols, c = i % cols;
    dst[(size_t)c * rows + r] = (__bf16)src[(size_t)r * cols + c];
  }
}

// ---------------------------------------------------------------------------
// Kernel 2: fused QKV projection. 8 waves/block share one 16-col n-tile;
// the three weight panels are TDM-staged to LDS per 256-wide K chunk.
//   Qb[m][o] = (query . Wq)/8 ; Kb[m][o] = key . Wk ; Vt[b][n][h][t] = (key.Wv)^T
// ---------------------------------------------------------------------------
__global__ void __launch_bounds__(256) qkv_kernel(
    const float* __restrict__ Xq, const float* __restrict__ Xk,
    const __bf16* __restrict__ Wqt, const __bf16* __restrict__ Wkt,
    const __bf16* __restrict__ Wvt,
    __bf16* __restrict__ Qb, __bf16* __restrict__ Kb,
    __bf16* __restrict__ Vt) {
  __shared__ __align__(16) __bf16 sw[3][16 * KB];   // 24 KB

  int lane = threadIdx.x & 31;
  int wave = threadIdx.x >> 5;
  int nt = blockIdx.x >> 5;                  // 48 n-tiles
  int mg = blockIdx.x & 31;                  // 32 groups of 8 m-tiles
  int m0 = (mg * 8 + wave) * 16;
  int n0 = nt * 16;
  int half = lane >> 4, l16 = lane & 15;
  int arow = m0 + l16;

  v8f cq = {}, ck = {}, cv = {};
  for (int kb = 0; kb < Dn; kb += KB) {
#if USE_TDM
    if (wave == 0) {
      unsigned base = (unsigned)(uintptr_t)(void*)&sw[0][0];
      tdm_load_panel(Wqt + (size_t)n0 * Dn + kb, base);
      tdm_load_panel(Wkt + (size_t)n0 * Dn + kb, base + 16 * KB * 2);
      tdm_load_panel(Wvt + (size_t)n0 * Dn + kb, base + 2 * 16 * KB * 2);
      __builtin_amdgcn_s_wait_tensorcnt(0);
    }
#else
    stage_panel_coop(Wqt + (size_t)n0 * Dn + kb, sw[0], Dn);
    stage_panel_coop(Wkt + (size_t)n0 * Dn + kb, sw[1], Dn);
    stage_panel_coop(Wvt + (size_t)n0 * Dn + kb, sw[2], Dn);
#endif
    __syncthreads();

#pragma unroll 2
    for (int kk = 0; kk < KB; kk += 32) {
      v16bf aq = a_frag_f32(Xq, Dn, arow, kb + kk, half);
      v16bf ak = a_frag_f32(Xk, Dn, arow, kb + kk, half);
      v16bf bq = b_frag_lds(sw[0], l16, kk, half);
      v16bf bk = b_frag_lds(sw[1], l16, kk, half);
      v16bf bv = b_frag_lds(sw[2], l16, kk, half);
      cq = WMMA_BF16(aq, bq, cq);
      ck = WMMA_BF16(ak, bk, ck);
      cv = WMMA_BF16(ak, bv, cv);
    }
    __syncthreads();
  }

  // C layout: element r lives at (row m0 + r + 8*half, col n0 + l16)
#pragma unroll
  for (int r = 0; r < 8; ++r) {
    int m = m0 + r + 8 * half;
    int o = n0 + l16;
    Qb[(size_t)m * NK + o] = (__bf16)(cq[r] * 0.125f);  // 1/sqrt(64)
    Kb[(size_t)m * NK + o] = (__bf16)ck[r];
    int b = m >> 11, t = m & (Tn - 1);
    int hn = o >> 6, hh = o & (Hd - 1);
    Vt[((size_t)(b * Nh + hn) * Hd + hh) * Tn + t] = (__bf16)cv[r];
  }
}

// ---------------------------------------------------------------------------
// Kernel 3: flash attention. One wave per (b, head, 16-row f-tile).
// Online softmax over 32-wide t-chunks; P relaid C->A through LDS.
// ---------------------------------------------------------------------------
__global__ void __launch_bounds__(256) attn_kernel(
    const __bf16* __restrict__ Qb, const __bf16* __restrict__ Kb,
    const __bf16* __restrict__ Vt, __bf16* __restrict__ Cc) {
  __shared__ __align__(16) __bf16 lds[8][16 * 32];

  int lane = threadIdx.x & 31;
  int wave = threadIdx.x >> 5;
  int w = blockIdx.x * 8 + wave;             // 2*12*128 = 3072 waves
  int ft = w & 127;
  int n  = (w >> 7) % Nh;
  int b  = w / (128 * Nh);
  int f0 = ft * 16;
  int half = lane >> 4, l16 = lane & 15;

  const __bf16* qbase = Qb + (size_t)(b * Tn) * NK + n * Hd;
  const __bf16* kbase = Kb + (size_t)(b * Tn) * NK + n * Hd;  // cols over t
  const __bf16* vbase = Vt + (size_t)(b * Nh + n) * Hd * Tn;  // cols over h

  int qrow = f0 + l16;
  v16bf qa0 = a_frag_bf16(qbase, NK, qrow, 0, half);
  v16bf qa1 = a_frag_bf16(qbase, NK, qrow, 32, half);

  float mrow[8], srow[8];
#pragma unroll
  for (int r = 0; r < 8; ++r) { mrow[r] = -1e30f; srow[r] = 0.0f; }
  v8f o0 = {}, o1 = {}, o2 = {}, o3 = {};

  for (int tc = 0; tc < Tn; tc += 32) {
    // ----- S = Q . K^T for columns tc..tc+31 (WMMA K dim = Hd = 64) -----
    v8f s0 = {}, s1 = {};
    {
      v16bf kb0 = b_frag_col(kbase, NK, tc + l16, 0, half);
      v16bf kb1 = b_frag_col(kbase, NK, tc + l16, 32, half);
      s0 = WMMA_BF16(qa0, kb0, s0);
      s0 = WMMA_BF16(qa1, kb1, s0);
      v16bf kb2 = b_frag_col(kbase, NK, tc + 16 + l16, 0, half);
      v16bf kb3 = b_frag_col(kbase, NK, tc + 16 + l16, 32, half);
      s1 = WMMA_BF16(qa0, kb2, s1);
      s1 = WMMA_BF16(qa1, kb3, s1);
    }

    // ----- online softmax; row r spans one 16-lane half -----
#pragma unroll
    for (int r = 0; r < 8; ++r) {
      float mx = fmaxf(s0[r], s1[r]);
#pragma unroll
      for (int msk = 8; msk >= 1; msk >>= 1)
        mx = fmaxf(mx, __shfl_xor(mx, msk, 32));
      float mnew  = fmaxf(mrow[r], mx);
      float alpha = __expf(mrow[r] - mnew);
      float p0 = __expf(s0[r] - mnew);
      float p1 = __expf(s1[r] - mnew);
      float ps = p0 + p1;
#pragma unroll
      for (int msk = 8; msk >= 1; msk >>= 1)
        ps += __shfl_xor(ps, msk, 32);
      srow[r] = srow[r] * alpha + ps;
      mrow[r] = mnew;
      o0[r] *= alpha; o1[r] *= alpha; o2[r] *= alpha; o3[r] *= alpha;
      s0[r] = p0; s1[r] = p1;
    }

    // ----- stage P (16x32 bf16) to LDS row-major, reload as A frag -----
    __bf16* lp = lds[wave];
#pragma unroll
    for (int r = 0; r < 8; ++r) {
      int row = r + 8 * half;
      lp[row * 32 + l16]      = (__bf16)s0[r];
      lp[row * 32 + 16 + l16] = (__bf16)s1[r];
    }
    // same-wave DS ops are in order; compiler inserts s_wait_dscnt
    v16bf pa;
    {
      const __bf16* p = lp + l16 * 32 + half * 8;
      v8bf lo = *(const v8bf*)(p);
      v8bf hi = *(const v8bf*)(p + 16);
      pa = __builtin_shufflevector(lo, hi, 0, 1, 2, 3, 4, 5, 6, 7,
                                           8, 9, 10, 11, 12, 13, 14, 15);
    }

    // ----- O += P . V  (K dim = 32 t-values; 4 h-tiles of 16) -----
    v16bf vb0 = b_frag_col(vbase, Tn,  0 + l16, tc, half);
    v16bf vb1 = b_frag_col(vbase, Tn, 16 + l16, tc, half);
    v16bf vb2 = b_frag_col(vbase, Tn, 32 + l16, tc, half);
    v16bf vb3 = b_frag_col(vbase, Tn, 48 + l16, tc, half);
    o0 = WMMA_BF16(pa, vb0, o0);
    o1 = WMMA_BF16(pa, vb1, o1);
    o2 = WMMA_BF16(pa, vb2, o2);
    o3 = WMMA_BF16(pa, vb3, o3);
  }

  // ----- normalize and store concated [b][f][n*64+h] as bf16 -----
#pragma unroll
  for (int r = 0; r < 8; ++r) {
    int f = f0 + r + 8 * half;
    float inv = 1.0f / srow[r];
    size_t base = (size_t)(b * Tn + f) * NK + n * Hd + l16;
    Cc[base +  0] = (__bf16)(o0[r] * inv);
    Cc[base + 16] = (__bf16)(o1[r] * inv);
    Cc[base + 32] = (__bf16)(o2[r] * inv);
    Cc[base + 48] = (__bf16)(o3[r] * inv);
  }
}

// ---------------------------------------------------------------------------
// Kernel 4: output projection, Wp panel TDM-staged to LDS per K chunk.
// out[m][d] = concated . Wp  (f32 out)
// ---------------------------------------------------------------------------
__global__ void __launch_bounds__(256) proj_kernel(
    const __bf16* __restrict__ Cc, const __bf16* __restrict__ Wpt,
    float* __restrict__ out) {
  __shared__ __align__(16) __bf16 sw[16 * KB];      // 8 KB

  int lane = threadIdx.x & 31;
  int wave = threadIdx.x >> 5;
  int nt = blockIdx.x >> 5;
  int mg = blockIdx.x & 31;
  int m0 = (mg * 8 + wave) * 16;
  int n0 = nt * 16;
  int half = lane >> 4, l16 = lane & 15;
  int arow = m0 + l16;

  v8f c = {};
  for (int kb = 0; kb < NK; kb += KB) {
#if USE_TDM
    if (wave == 0) {
      tdm_load_panel(Wpt + (size_t)n0 * NK + kb,
                     (unsigned)(uintptr_t)(void*)&sw[0]);
      __builtin_amdgcn_s_wait_tensorcnt(0);
    }
#else
    stage_panel_coop(Wpt + (size_t)n0 * NK + kb, sw, NK);
#endif
    __syncthreads();

#pragma unroll 2
    for (int kk = 0; kk < KB; kk += 32) {
      v16bf a  = a_frag_bf16(Cc, NK, arow, kb + kk, half);
      v16bf bw = b_frag_lds(sw, l16, kk, half);
      c = WMMA_BF16(a, bw, c);
    }
    __syncthreads();
  }

#pragma unroll
  for (int r = 0; r < 8; ++r) {
    int m = m0 + r + 8 * half;
    out[(size_t)m * Dn + n0 + l16] = c[r];
  }
}

// ---------------------------------------------------------------------------
// Host launcher
// ---------------------------------------------------------------------------
extern "C" void kernel_launch(void* const* d_in, const int* in_sizes, int n_in,
                              void* d_out, int out_size, void* d_ws,
                              size_t ws_size, hipStream_t stream) {
  const float* q  = (const float*)d_in[0];
  const float* k  = (const float*)d_in[1];
  const float* Wq = (const float*)d_in[2];
  const float* Wk = (const float*)d_in[3];
  const float* Wv = (const float*)d_in[4];
  const float* Wp = (const float*)d_in[5];
  float* out = (float*)d_out;

  // workspace carve-up (bf16 buffers)
  const size_t wbytes = (size_t)Dn * NK * sizeof(__bf16);   // 1.125 MB each
  const size_t abytes = (size_t)Mn * NK * sizeof(__bf16);   // 6 MB each
  char* ws = (char*)d_ws;
  __bf16* Wqt = (__bf16*)(ws);                 ws += wbytes;
  __bf16* Wkt = (__bf16*)(ws);                 ws += wbytes;
  __bf16* Wvt = (__bf16*)(ws);                 ws += wbytes;
  __bf16* Wpt = (__bf16*)(ws);                 ws += wbytes;
  __bf16* Qb  = (__bf16*)(ws);                 ws += abytes;
  __bf16* Kb  = (__bf16*)(ws);                 ws += abytes;
  __bf16* Vt  = (__bf16*)(ws);                 ws += abytes;
  __bf16* Cc  = (__bf16*)(ws);                 ws += abytes;

  // 1) weights -> bf16, transposed to [out][K]
  int wn = Dn * NK;
  transpose_bf16<<<(wn + 255) / 256, 256, 0, stream>>>(Wq, Wqt, Dn, NK);
  transpose_bf16<<<(wn + 255) / 256, 256, 0, stream>>>(Wk, Wkt, Dn, NK);
  transpose_bf16<<<(wn + 255) / 256, 256, 0, stream>>>(Wv, Wvt, Dn, NK);
  transpose_bf16<<<(wn + 255) / 256, 256, 0, stream>>>(Wp, Wpt, NK, Dn);

  // 2) fused QKV projection: 48 n-tiles x 32 m-groups, 8 waves/block
  qkv_kernel<<<48 * 32, 256, 0, stream>>>(q, k, Wqt, Wkt, Wvt, Qb, Kb, Vt);

  // 3) flash attention: 2*12*128 waves, 8/block
  attn_kernel<<<Bn * Nh * (Tn / 16) / 8, 256, 0, stream>>>(Qb, Kb, Vt, Cc);

  // 4) output projection
  proj_kernel<<<48 * 32, 256, 0, stream>>>(Cc, Wpt, out);
}
